// CrystalGraphConvNet_43095701848252
// MI455X (gfx1250) — matrix operand order repaired and checked by reading, not compiled, not run on hardware
//
#include <hip/hip_runtime.h>
#include <math.h>

// CGCNN forward for MI455X (gfx1250, wave32).
// Dominant GEMM (1.2M x 169) @ (169 x 128) via v_wmma_f32_16x16x32_f16.
// BN statistics fused into the GEMM epilogue (two-pass recompute) -> the
// 614MB/layer intermediate "g" tensor is never written to HBM.
// Operands staged in f16: x_h (12.8MB) and padded nbr_fea_h (98MB, L2-resident
// across all 6 GEMM passes on the 192MB L2) -> A-fragments are two aligned
// 16B loads per K-chunk, no cvt in the hot loop.

typedef __attribute__((ext_vector_type(16))) _Float16 v16h;
typedef __attribute__((ext_vector_type(8)))  _Float16 v8h;
typedef __attribute__((ext_vector_type(8)))  float    v8f;

#define F_DIM   64
#define TWOF    128
#define NBR_D   41
#define NBR_P   48      // nbr_fea padded row stride (halves) -> clean 8-half runs
#define KTOT    169     // 2F + NBR
#define KCHUNKS 6       // 6 * 32 = 192 >= 169 (zero padded)
#define NTILES  8       // 128 cols / 16
#define M_NBR   12
#define ORIG_D  92

__device__ __forceinline__ float softplusf_(float v) {
    return v > 20.0f ? v : log1pf(expf(v));
}
__device__ __forceinline__ float sigmoidf_(float v) {
    return 1.0f / (1.0f + expf(-v));
}
__device__ __forceinline__ v8h zero8() {
    v8h z = {(_Float16)0.0f, (_Float16)0.0f, (_Float16)0.0f, (_Float16)0.0f,
             (_Float16)0.0f, (_Float16)0.0f, (_Float16)0.0f, (_Float16)0.0f};
    return z;
}

// ---------------------------------------------------------------------------
// Pack fcW (3 x 169 x 128, f32) into WMMA B-fragment lane layout, f16.
// packB[((l*6 + kc)*8 + nt)*512 + lane*16 + h]
//   k = kc*32 + (lane>>4)*8 + (h&7) + (h>>3)*16    (B 32x16: table order, matches A)
//   n = nt*16 + (lane&15)
// ---------------------------------------------------------------------------
__global__ void cg_pack_b(const float* __restrict__ fcW, _Float16* __restrict__ packB,
                          int total) {
    int gid = blockIdx.x * blockDim.x + threadIdx.x;
    if (gid >= total) return;
    int h    = gid & 15;
    int lane = (gid >> 4) & 31;
    int nt   = (gid >> 9) & 7;
    int rem  = gid >> 12;
    int kc   = rem % KCHUNKS;
    int l    = rem / KCHUNKS;
    int k    = kc * 32 + ((lane >> 4) << 3) + (h & 7) + ((h >> 3) << 4);
    int n    = nt * 16 + (lane & 15);
    float v  = (k < KTOT) ? fcW[(l * KTOT + k) * TWOF + n] : 0.0f;
    packB[gid] = (_Float16)v;
}

// ---------------------------------------------------------------------------
// Convert nbr_fea (nRows x 41 f32) -> padded f16 (nRows x 48), zero padded.
// ---------------------------------------------------------------------------
__global__ void cg_pack_nf(const float* __restrict__ nf, _Float16* __restrict__ nfh,
                           int nRows) {
    int i = blockIdx.x * blockDim.x + threadIdx.x;
    if (i >= nRows * NBR_P) return;
    int row = i / NBR_P;
    int j   = i - row * NBR_P;
    nfh[i] = (_Float16)((j < NBR_D) ? nf[(size_t)row * NBR_D + j] : 0.0f);
}

// ---------------------------------------------------------------------------
// Embedding: x = (atom_fea * mask) @ embW + embb  (also writes f16 mirror)
// ---------------------------------------------------------------------------
__global__ void cg_embed(const float* __restrict__ af, const float* __restrict__ mask,
                         const float* __restrict__ w, const float* __restrict__ b,
                         float* __restrict__ x, _Float16* __restrict__ xh, int n) {
    int i = blockIdx.x * blockDim.x + threadIdx.x;
    if (i >= n * F_DIM) return;
    int row = i >> 6;
    int f   = i & 63;
    float acc = b[f];
    #pragma unroll 4
    for (int k = 0; k < ORIG_D; ++k)
        acc += af[row * ORIG_D + k] * mask[k] * w[k * F_DIM + f];
    x[i]  = acc;
    xh[i] = (_Float16)acc;
}

// ---------------------------------------------------------------------------
// Load one 8-half A-run starting at (8-aligned) k of the concatenated row
// [x_self(64) | x_nbr(64) | nbr_fea(41, pad to 48) | zeros].
// ---------------------------------------------------------------------------
__device__ __forceinline__ v8h load_run8(const _Float16* __restrict__ xs,
                                         const _Float16* __restrict__ xn,
                                         const _Float16* __restrict__ nfh,
                                         const float* __restrict__ nf32, int k) {
    if (k < F_DIM)     return *(const v8h*)(xs + k);
    if (k < 2 * F_DIM) return *(const v8h*)(xn + (k - F_DIM));
    v8h r = zero8();
    if (k < 2 * F_DIM + NBR_P) {
        if (nfh) {
            r = *(const v8h*)(nfh + (k - 2 * F_DIM));
        } else {
            #pragma unroll
            for (int j = 0; j < 8; ++j) {
                int kk = k - 2 * F_DIM + j;
                r[j] = (_Float16)((kk < NBR_D) ? nf32[kk] : 0.0f);
            }
        }
    }
    return r;
}

// ---------------------------------------------------------------------------
// Fused conv GEMM.  One wave = one 16-row block of tot (rows of N*M).
// pass 0: per-channel sum / sumsq of d = tot @ W (bias cancels inside BN)
// pass 1: g = d*scale + shift; summed[atom][c] += sigmoid(g_c)*softplus(g_{c+64})
// 4 waves / block -> 64 rows per block.
// ---------------------------------------------------------------------------
__global__ __launch_bounds__(128)
void cg_conv_gemm(const _Float16* __restrict__ xh,
                  const int* __restrict__ nbr_idx,
                  const _Float16* __restrict__ nfh,     // padded f16, or null
                  const float* __restrict__ nf32,       // fallback fp32 nbr_fea
                  const _Float16* __restrict__ packB,   // this layer, 6*8*512 halves
                  float* __restrict__ stats,            // pass0: [sum(128), sumsq(128)]
                  const float* __restrict__ scale,      // pass1: 128
                  const float* __restrict__ shift,      // pass1: 128
                  float* __restrict__ summed,           // pass1: N*64, atomic
                  int pass, int nRows) {
    __shared__ float sStat[2 * TWOF];
    if (pass == 0) {
        for (int i = threadIdx.x; i < 2 * TWOF; i += blockDim.x) sStat[i] = 0.0f;
        __syncthreads();
    }

    const int lane = threadIdx.x & 31;
    const int wid  = threadIdx.x >> 5;
    const int g    = lane >> 4;     // lane group
    const int ln   = lane & 15;
    const int r0   = (blockIdx.x * 4 + wid) * 16;

    if (r0 < nRows) {
        int grow = r0 + ln;
        if (grow >= nRows) grow = nRows - 1;
        const int a            = grow / M_NBR;
        const _Float16* xs     = xh + (size_t)a * F_DIM;
        const _Float16* xn     = xh + (size_t)nbr_idx[grow] * F_DIM;
        const _Float16* nfrow  = nfh ? nfh + (size_t)grow * NBR_P : (const _Float16*)0;
        const float*    nfrow32= nf32 + (size_t)grow * NBR_D;

        v8f acc[NTILES];
        const v8f vzero = {0.f, 0.f, 0.f, 0.f, 0.f, 0.f, 0.f, 0.f};
        #pragma unroll
        for (int t = 0; t < NTILES; ++t) acc[t] = vzero;

        #pragma unroll
        for (int kc = 0; kc < KCHUNKS; ++kc) {
            // 16-bit 16x32 A layout: halves 0..7 -> K = kc*32 + g*8 + h,
            //                        halves 8..15 -> K = kc*32 + 16 + g*8 + (h-8)
            const int k0 = kc * 32 + (g << 3);
            union { v16h v; v8h h[2]; } af;
            af.h[0] = load_run8(xs, xn, nfrow, nfrow32, k0);
            af.h[1] = load_run8(xs, xn, nfrow, nfrow32, k0 + 16);

            const v16h* bp = (const v16h*)(packB + (size_t)(kc * NTILES) * 512 + lane * 16);
            #pragma unroll
            for (int nt = 0; nt < NTILES; ++nt) {
                v16h bv = bp[nt * 32];   // tiles are 512 halves = 32 v16h apart
                acc[nt] = __builtin_amdgcn_wmma_f32_16x16x32_f16(
                    false, af.v, false, bv, (short)0, acc[nt], false, false);
            }
        }

        // D layout: element r of v8f -> row m = r + 8*g, col = nt*16 + ln.
        if (pass == 0) {
            #pragma unroll
            for (int nt = 0; nt < NTILES; ++nt) {
                int col = nt * 16 + ln;
                float s = 0.0f, ss = 0.0f;
                #pragma unroll
                for (int r = 0; r < 8; ++r) {
                    int rr = r0 + r + 8 * g;
                    float d = acc[nt][r];
                    if (rr < nRows) { s += d; ss += d * d; }
                }
                atomicAdd(&sStat[col], s);
                atomicAdd(&sStat[TWOF + col], ss);
            }
        } else {
            #pragma unroll
            for (int nt = 0; nt < 4; ++nt) {
                int colf = nt * 16 + ln;       // filter channel
                int colc = colf + F_DIM;       // core channel
                float scf = scale[colf], shf = shift[colf];
                float scc = scale[colc], shc = shift[colc];
                #pragma unroll
                for (int r = 0; r < 8; ++r) {
                    int rr = r0 + r + 8 * g;
                    if (rr < nRows) {
                        float gf = acc[nt][r]     * scf + shf;
                        float gc = acc[nt + 4][r] * scc + shc;
                        float val = sigmoidf_(gf) * softplusf_(gc);
                        int atom = rr / M_NBR;
                        atomicAdd(&summed[(size_t)atom * F_DIM + colf], val);
                    }
                }
            }
        }
    }

    if (pass == 0) {
        __syncthreads();
        for (int i = threadIdx.x; i < 2 * TWOF; i += blockDim.x)
            atomicAdd(&stats[i], sStat[i]);
    }
}

// ---------------------------------------------------------------------------
// BN finalize: scale = gamma * rsqrt(var + eps), shift = beta - mean * scale
// ---------------------------------------------------------------------------
__global__ void cg_finalize_bn(const float* __restrict__ stats,
                               const float* __restrict__ gamma,
                               const float* __restrict__ beta,
                               float* __restrict__ scale, float* __restrict__ shift,
                               int nch, float invCnt) {
    int c = blockIdx.x * blockDim.x + threadIdx.x;
    if (c >= nch) return;
    float mean = stats[c] * invCnt;
    float var  = stats[nch + c] * invCnt - mean * mean;
    float inv  = rsqrtf(var + 1e-5f);
    float sc   = gamma[c] * inv;
    scale[c]   = sc;
    shift[c]   = beta[c] - mean * sc;
}

// ---------------------------------------------------------------------------
// Per-channel sum/sumsq of summed (N x 64) for bn2.
// ---------------------------------------------------------------------------
__global__ void cg_bn2_stats(const float* __restrict__ summed,
                             float* __restrict__ stats2, int n) {
    int t = blockIdx.x * blockDim.x + threadIdx.x;
    int f = t & 63;
    int row0 = t >> 6;
    int stride = (gridDim.x * blockDim.x) >> 6;
    float s = 0.0f, ss = 0.0f;
    for (int r = row0; r < n; r += stride) {
        float v = summed[(size_t)r * F_DIM + f];
        s += v; ss += v * v;
    }
    atomicAdd(&stats2[f], s);
    atomicAdd(&stats2[F_DIM + f], ss);
}

// ---------------------------------------------------------------------------
// x = softplus(x + bn2(summed))  (also refresh f16 mirror)
// ---------------------------------------------------------------------------
__global__ void cg_update_x(float* __restrict__ x, _Float16* __restrict__ xh,
                            const float* __restrict__ summed,
                            const float* __restrict__ scale,
                            const float* __restrict__ shift, int total) {
    int i = blockIdx.x * blockDim.x + threadIdx.x;
    if (i >= total) return;
    int f = i & 63;
    float v = softplusf_(x[i] + summed[i] * scale[f] + shift[f]);
    x[i]  = v;
    xh[i] = (_Float16)v;
}

__global__ void cg_zero(float* __restrict__ p, int n) {
    int i = blockIdx.x * blockDim.x + threadIdx.x;
    if (i < n) p[i] = 0.0f;
}

// ---------------------------------------------------------------------------
// Segment pooling: crysSum[seg][f] += x[n][f];  crysCnt[seg] += 1
// ---------------------------------------------------------------------------
__global__ void cg_pool(const float* __restrict__ x, const int* __restrict__ seg,
                        float* __restrict__ sums, float* __restrict__ cnt, int n) {
    int i = blockIdx.x * blockDim.x + threadIdx.x;
    if (i >= n * F_DIM) return;
    int row = i >> 6;
    int f   = i & 63;
    int s   = seg[row];
    atomicAdd(&sums[(size_t)s * F_DIM + f], x[i]);
    if (f == 0) atomicAdd(&cnt[s], 1.0f);
}

// ---------------------------------------------------------------------------
// Head: out[c] = softplus(softplus(mean_c) @ c2fW + c2fb) @ outW + outb
// One block (128 threads) per crystal.
// ---------------------------------------------------------------------------
__global__ __launch_bounds__(128)
void cg_head(const float* __restrict__ sums, const float* __restrict__ cnt,
             const float* __restrict__ c2fW, const float* __restrict__ c2fb,
             const float* __restrict__ outW, const float* __restrict__ outb,
             float* __restrict__ out) {
    __shared__ float sMean[F_DIM];
    __shared__ float sRed[128];
    int c = blockIdx.x;
    int t = threadIdx.x;
    if (t < F_DIM) {
        float d = fmaxf(cnt[c], 1.0f);
        sMean[t] = softplusf_(sums[(size_t)c * F_DIM + t] / d);
    }
    __syncthreads();
    float h = c2fb[t];
    #pragma unroll 4
    for (int f = 0; f < F_DIM; ++f) h += sMean[f] * c2fW[f * 128 + t];
    h = softplusf_(h);
    sRed[t] = h * outW[t];
    __syncthreads();
    for (int sdiv = 64; sdiv > 0; sdiv >>= 1) {
        if (t < sdiv) sRed[t] += sRed[t + sdiv];
        __syncthreads();
    }
    if (t == 0) out[c] = sRed[0] + outb[0];
}

// ---------------------------------------------------------------------------
extern "C" void kernel_launch(void* const* d_in, const int* in_sizes, int n_in,
                              void* d_out, int out_size, void* d_ws, size_t ws_size,
                              hipStream_t stream) {
    const float* atom_fea = (const float*)d_in[0];
    const float* nbr_fea  = (const float*)d_in[1];
    const int*   nbr_idx  = (const int*)d_in[2];
    const int*   seg_ids  = (const int*)d_in[3];
    // d_in[4] = n_crystals (device scalar; C == out_size)
    const float* mask     = (const float*)d_in[5];
    const float* embW     = (const float*)d_in[6];
    const float* embb     = (const float*)d_in[7];
    const float* fcW      = (const float*)d_in[8];
    // d_in[9] = fcb: cancels analytically inside batch-norm, unused.
    const float* bn1g     = (const float*)d_in[10];
    const float* bn1b     = (const float*)d_in[11];
    const float* bn2g     = (const float*)d_in[12];
    const float* bn2b     = (const float*)d_in[13];
    const float* c2fW     = (const float*)d_in[14];
    const float* c2fb     = (const float*)d_in[15];
    const float* outW     = (const float*)d_in[16];
    const float* outb     = (const float*)d_in[17];
    float* out = (float*)d_out;

    const int N     = in_sizes[0] / ORIG_D;
    const int Mn    = in_sizes[2] / N;
    const int C     = out_size;
    const int nRows = N * Mn;

    // Workspace carving (all offsets 256B aligned; >=16B alignment for v8h/v16h).
    char* wsp = (char*)d_ws;
    size_t used = 0;
    auto carve = [&](size_t bytes) -> char* {
        char* p = wsp + used;
        used += (bytes + 255) & ~(size_t)255;
        return p;
    };
    float*    x       = (float*)carve((size_t)N * F_DIM * 4);
    float*    summed  = (float*)carve((size_t)N * F_DIM * 4);
    _Float16* xh      = (_Float16*)carve((size_t)N * F_DIM * 2);
    _Float16* packB   = (_Float16*)carve((size_t)3 * KCHUNKS * NTILES * 512 * 2);
    float*    stats1  = (float*)carve(2 * TWOF * 4);
    float*    scale1  = (float*)carve(TWOF * 4);
    float*    shift1  = (float*)carve(TWOF * 4);
    float*    stats2  = (float*)carve(2 * F_DIM * 4);
    float*    scale2  = (float*)carve(F_DIM * 4);
    float*    shift2  = (float*)carve(F_DIM * 4);
    float*    crysSum = (float*)carve((size_t)C * F_DIM * 4);
    float*    crysCnt = (float*)carve((size_t)C * 4);
    // Optional f16 staging of nbr_fea (98MB @ padded 48 halves/row): carved last,
    // only if workspace allows. Host-side fixed decision -> deterministic.
    _Float16* nfh = (_Float16*)0;
    {
        size_t nfhBytes = (size_t)nRows * NBR_P * 2;
        if (used + nfhBytes + 256 <= ws_size)
            nfh = (_Float16*)carve(nfhBytes);
    }
    (void)n_in;

    const int packTotal = 3 * KCHUNKS * NTILES * 512;   // 73728
    cg_pack_b<<<(packTotal + 255) / 256, 256, 0, stream>>>(fcW, packB, packTotal);
    if (nfh)
        cg_pack_nf<<<(int)(((size_t)nRows * NBR_P + 255) / 256), 256, 0, stream>>>(
            nbr_fea, nfh, nRows);

    cg_embed<<<(N * F_DIM + 255) / 256, 256, 0, stream>>>(atom_fea, mask, embW, embb,
                                                          x, xh, N);

    const int gemmGrid = (nRows + 63) / 64;   // 4 waves * 16 rows per block
    for (int l = 0; l < 3; ++l) {
        const _Float16* Bl = packB + (size_t)l * KCHUNKS * NTILES * 512;

        cg_zero<<<1, 256, 0, stream>>>(stats1, 2 * TWOF);
        cg_conv_gemm<<<gemmGrid, 128, 0, stream>>>(xh, nbr_idx, nfh, nbr_fea, Bl,
                                                   stats1, nullptr, nullptr, nullptr,
                                                   0, nRows);
        cg_finalize_bn<<<1, 128, 0, stream>>>(stats1, bn1g + l * TWOF, bn1b + l * TWOF,
                                              scale1, shift1, TWOF, 1.0f / (float)nRows);

        cg_zero<<<(N * F_DIM + 255) / 256, 256, 0, stream>>>(summed, N * F_DIM);
        cg_conv_gemm<<<gemmGrid, 128, 0, stream>>>(xh, nbr_idx, nfh, nbr_fea, Bl,
                                                   nullptr, scale1, shift1, summed,
                                                   1, nRows);

        cg_zero<<<1, 128, 0, stream>>>(stats2, 2 * F_DIM);
        cg_bn2_stats<<<64, 256, 0, stream>>>(summed, stats2, N);
        cg_finalize_bn<<<1, 64, 0, stream>>>(stats2, bn2g + l * F_DIM, bn2b + l * F_DIM,
                                             scale2, shift2, F_DIM, 1.0f / (float)N);
        cg_update_x<<<(N * F_DIM + 255) / 256, 256, 0, stream>>>(x, xh, summed, scale2,
                                                                 shift2, N * F_DIM);
    }

    cg_zero<<<(C * F_DIM + 255) / 256, 256, 0, stream>>>(crysSum, C * F_DIM);
    cg_zero<<<(C + 255) / 256, 256, 0, stream>>>(crysCnt, C);
    cg_pool<<<(N * F_DIM + 255) / 256, 256, 0, stream>>>(x, seg_ids, crysSum, crysCnt, N);
    cg_head<<<C, 128, 0, stream>>>(crysSum, crysCnt, c2fW, c2fb, outW, outb, out);
}
